// AttentionLayer_4999341933070
// MI455X (gfx1250) — matrix operand order repaired and checked by reading, not compiled, not run on hardware
//
#include <hip/hip_runtime.h>

#define B_    4
#define CIN   128
#define NSEQ  4096
#define CKEY  64
#define COUT  128

typedef __attribute__((ext_vector_type(16))) __bf16 v16bf;
typedef __attribute__((ext_vector_type(8)))  float  v8f;
typedef unsigned int uint4v __attribute__((ext_vector_type(4)));

union Frag16 {
    v16bf  v;
    uint4v q[2];
};

// hardware f32 -> bf16 (RNE) conversion
__device__ __forceinline__ unsigned short f32_to_bf16(float f) {
    __bf16 h = (__bf16)f;
    return __builtin_bit_cast(unsigned short, h);
}

#define WMMA_BF16(A, Bm, C) \
    __builtin_amdgcn_wmma_f32_16x16x32_bf16(false, (A), false, (Bm), (short)0, (C), false, false)

// gfx1250 async DMA: copy 16 bytes global -> LDS per lane, tracked by ASYNCcnt.
__device__ __forceinline__ void async_b128(unsigned lds_off, const void* gaddr) {
    asm volatile("global_load_async_to_lds_b128 %0, %1, off"
                 :: "v"(lds_off), "v"(gaddr) : "memory");
}
__device__ __forceinline__ void wait_async0() {
    asm volatile("s_wait_asynccnt 0" ::: "memory");
}

// butterfly step via ds_swizzle_b32 immediate XOR pattern (no index VGPR setup)
template <int MASK>
__device__ __forceinline__ float swz_xor(float v) {
    int j = __builtin_amdgcn_ds_swizzle(__builtin_bit_cast(int, v), (MASK << 10) | 0x1f);
    return __builtin_bit_cast(float, j);
}
__device__ __forceinline__ float red_max16(float v) {
    v = fmaxf(v, swz_xor<1>(v));
    v = fmaxf(v, swz_xor<2>(v));
    v = fmaxf(v, swz_xor<4>(v));
    v = fmaxf(v, swz_xor<8>(v));
    return v;
}
__device__ __forceinline__ float red_sum16(float v) {
    v += swz_xor<1>(v);
    v += swz_xor<2>(v);
    v += swz_xor<4>(v);
    v += swz_xor<8>(v);
    return v;
}

// ---------------------------------------------------------------------------
// Kernel 1: 1x1-conv projections, f32 -> bf16, into WMMA-friendly layouts.
//   Qb[b][n][c] = Q^T   (c in 0..63)
//   Kb[b][n][c] = K^T   (c in 0..63)
//   Vb[b][c][n] = V     (c in 0..127)
// ---------------------------------------------------------------------------
__global__ __launch_bounds__(256)
void proj_kernel(const float* __restrict__ x,
                 const float* __restrict__ Wq,
                 const float* __restrict__ Wk,
                 const float* __restrict__ Wv,
                 unsigned short* __restrict__ Qb,
                 unsigned short* __restrict__ Kb,
                 unsigned short* __restrict__ Vb) {
    __shared__ float xs[CIN * 32];   // x[b, :, n0..n0+31] tile, 16 KB

    const int nt = NSEQ / 32;
    const int b  = blockIdx.x / nt;
    const int n0 = (blockIdx.x % nt) * 32;
    const int t  = threadIdx.x;

#pragma unroll
    for (int r = 0; r < 16; ++r) {
        int idx = t + r * 256;
        int i = idx >> 5, n = idx & 31;
        xs[idx] = x[((size_t)b * CIN + i) * NSEQ + n0 + n];
    }
    __syncthreads();

    const int c = t;   // 0..63 Q, 64..127 K, 128..255 V
    const float* Wrow = (c < 64)  ? (Wq + (size_t)c * CIN)
                      : (c < 128) ? (Wk + (size_t)(c - 64) * CIN)
                                  : (Wv + (size_t)(c - 128) * CIN);
    float acc[32];
#pragma unroll
    for (int n = 0; n < 32; ++n) acc[n] = 0.0f;

    for (int i = 0; i < CIN; ++i) {
        float w = Wrow[i];
        const float* xr = &xs[i * 32];
#pragma unroll
        for (int n = 0; n < 32; ++n) acc[n] = fmaf(w, xr[n], acc[n]);
    }

    if (c < 64) {
        unsigned short* q = Qb + ((size_t)b * NSEQ + n0) * CKEY + c;
#pragma unroll
        for (int n = 0; n < 32; ++n) q[(size_t)n * CKEY] = f32_to_bf16(acc[n]);
    } else if (c < 128) {
        unsigned short* k = Kb + ((size_t)b * NSEQ + n0) * CKEY + (c - 64);
#pragma unroll
        for (int n = 0; n < 32; ++n) k[(size_t)n * CKEY] = f32_to_bf16(acc[n]);
    } else {
        unsigned short* v = Vb + ((size_t)b * COUT + (c - 128)) * NSEQ + n0;
#pragma unroll
        for (int n = 0; n < 32; ++n) v[n] = f32_to_bf16(acc[n]);
    }
}

// ---------------------------------------------------------------------------
// Kernel 2: fused flash attention, async-LDS double-buffered K/V staging.
//   block = 128 threads = 4 wave32; each wave owns 16 query rows (64/block).
//   Per 64-key chunk: DMA K (8KB) + V (16KB) to LDS, S = Q^T K (8 WMMA),
//   online softmax (deferred row-sum), O += P V^T (16 WMMA).
// ---------------------------------------------------------------------------
__global__ __launch_bounds__(128)
void attn_kernel(const unsigned short* __restrict__ Qb,
                 const unsigned short* __restrict__ Kb,
                 const unsigned short* __restrict__ Vb,
                 float* __restrict__ out) {
    __shared__ __attribute__((aligned(16))) unsigned short Kt[2][64 * 64];    // [key][ch]
    __shared__ __attribute__((aligned(16))) unsigned short Vt[2][COUT * 64];  // [ch][key]
    __shared__ __attribute__((aligned(16))) unsigned short Pt[4][16 * 64];    // per-wave P

    const int tiles = NSEQ / 64;
    const int b     = blockIdx.x / tiles;
    const int q0    = (blockIdx.x % tiles) * 64 + (int)(threadIdx.x >> 5) * 16;
    const int lane  = threadIdx.x & 31;
    const int wave  = threadIdx.x >> 5;
    const int hs    = lane >> 4;     // lane half (A-frag K split / C-frag row split)
    const int ln    = lane & 15;     // column / row-within-half index

    const unsigned short* Kc = Kb + (size_t)b * NSEQ * CKEY;   // batch K^T base
    const unsigned short* Vc = Vb + (size_t)b * COUT * NSEQ;   // batch V base

    // ---- issue one 64-key chunk's DMA (24 KB); 12 b128 wave-instructions/wave
    auto issue_chunk = [&](int kc, int buf) {
#pragma unroll
        for (int j = 0; j < 12; ++j) {
            const int t = wave * 12 + j;           // 0..47
            if (t < 16) {                          // K region: contiguous 8 KB
                unsigned o = (unsigned)t * 512u + (unsigned)lane * 16u;
                async_b128((unsigned)(unsigned long long)((const char*)&Kt[buf][0] + o),
                           (const char*)(Kc + (size_t)kc * CKEY) + o);
            } else {                               // V region: 128 rows x 128 B
                unsigned o  = (unsigned)(t - 16) * 512u + (unsigned)lane * 16u;
                unsigned ch = o >> 7;              // row = 64 keys * 2 B = 128 B
                unsigned w  = o & 127u;
                async_b128((unsigned)(unsigned long long)((const char*)&Vt[buf][0] + o),
                           (const char*)(Vc + ((size_t)ch * NSEQ + kc)) + w);
            }
        }
    };

    // ---- Q A-fragments: rows q0..q0+15, channels 0..63 (two K=32 fragments)
    Frag16 aq[2];
    const unsigned short* qp = Qb + ((size_t)b * NSEQ + q0 + ln) * CKEY;
#pragma unroll
    for (int j = 0; j < 2; ++j) {
        aq[j].q[0] = *(const uint4v*)(qp + j * 32 + hs * 8);
        aq[j].q[1] = *(const uint4v*)(qp + j * 32 + 16 + hs * 8);
    }

    v8f o[8];
#pragma unroll
    for (int ot = 0; ot < 8; ++ot) o[ot] = (v8f){};
    float m[8], lsum[8];                     // lsum = per-LANE partial row sum
#pragma unroll
    for (int g = 0; g < 8; ++g) { m[g] = -1e30f; lsum[g] = 0.0f; }

    issue_chunk(0, 0);                       // prologue DMA

    for (int kc = 0; kc < NSEQ; kc += 64) {
        const int bufi = (kc >> 6) & 1;

        wait_async0();                       // my chunk-kc transfers landed
        __syncthreads();                     // everyone's landed; buf^1 free
        if (kc + 64 < NSEQ)
            issue_chunk(kc + 64, bufi ^ 1);  // overlap next DMA with compute

        // ---- S = Q^T K : four 16x16 key tiles, K=64 channels in 2 WMMA steps
        v8f s[4];
        const unsigned short* ktile = &Kt[bufi][0];
#pragma unroll
        for (int kn = 0; kn < 4; ++kn) {
            const unsigned short* kp = ktile + ((kn * 16 + ln) * 64 + hs * 8);
            Frag16 bk0, bk1;
            bk0.q[0] = ((const uint4v*)kp)[0];
            bk0.q[1] = ((const uint4v*)kp)[2];
            bk1.q[0] = ((const uint4v*)(kp + 32))[0];
            bk1.q[1] = ((const uint4v*)(kp + 32))[2];
            v8f c = (v8f){};
            c = WMMA_BF16(aq[0].v, bk0.v, c);
            c = WMMA_BF16(aq[1].v, bk1.v, c);
            s[kn] = c;
        }

        // ---- online softmax: only the row-max needs a cross-lane reduction
        float scale[8];
#pragma unroll
        for (int g = 0; g < 8; ++g) {
            float tmax = fmaxf(fmaxf(s[0][g], s[1][g]), fmaxf(s[2][g], s[3][g]));
            tmax = red_max16(tmax);
            float mn = fmaxf(m[g], tmax);
            float sc = __expf(m[g] - mn);
            float p0 = __expf(s[0][g] - mn);
            float p1 = __expf(s[1][g] - mn);
            float p2 = __expf(s[2][g] - mn);
            float p3 = __expf(s[3][g] - mn);
            lsum[g] = lsum[g] * sc + ((p0 + p1) + (p2 + p3));  // per-lane partial
            m[g]     = mn;
            scale[g] = sc;
            unsigned short* pr = &Pt[wave][(g + 8 * hs) * 64 + ln];
            pr[0]  = f32_to_bf16(p0);
            pr[16] = f32_to_bf16(p1);
            pr[32] = f32_to_bf16(p2);
            pr[48] = f32_to_bf16(p3);
        }

        // rescale O accumulators
#pragma unroll
        for (int ot = 0; ot < 8; ++ot)
#pragma unroll
            for (int g = 0; g < 8; ++g)
                o[ot][g] = o[ot][g] * scale[g];

        asm volatile("s_wait_dscnt 0" ::: "memory");   // P stores visible

        // ---- P A-fragments (two K=32 key groups)
        Frag16 ap[2];
#pragma unroll
        for (int kq = 0; kq < 2; ++kq) {
            const unsigned short* pp = &Pt[wave][ln * 64 + kq * 32 + hs * 8];
            ap[kq].q[0] = *(const uint4v*)pp;
            ap[kq].q[1] = *(const uint4v*)(pp + 16);
        }

        // ---- O += P @ V^T : 8 output-channel tiles x 2 key groups
#pragma unroll
        for (int kq = 0; kq < 2; ++kq)
#pragma unroll
            for (int ot = 0; ot < 8; ++ot) {
                const unsigned short* vp =
                    &Vt[bufi][(ot * 16 + ln) * 64 + kq * 32 + hs * 8];
                Frag16 bv;
                bv.q[0] = ((const uint4v*)vp)[0];
                bv.q[1] = ((const uint4v*)vp)[2];
                o[ot] = WMMA_BF16(ap[kq].v, bv.v, o[ot]);
            }
    }

    // ---- epilogue: reduce per-lane partial row sums once, then divide+store
#pragma unroll
    for (int g = 0; g < 8; ++g)
        lsum[g] = 1.0f / red_sum16(lsum[g]);
#pragma unroll
    for (int ot = 0; ot < 8; ++ot) {
        int ch = ot * 16 + ln;
        float* op = out + ((size_t)b * COUT + ch) * NSEQ + q0;
#pragma unroll
        for (int g = 0; g < 8; ++g)
            op[g + 8 * hs] = o[ot][g] * lsum[g];
    }
}

// ---------------------------------------------------------------------------
extern "C" void kernel_launch(void* const* d_in, const int* in_sizes, int n_in,
                              void* d_out, int out_size, void* d_ws, size_t ws_size,
                              hipStream_t stream) {
    const float* x  = (const float*)d_in[0];
    const float* Wq = (const float*)d_in[1];
    const float* Wk = (const float*)d_in[2];
    const float* Wv = (const float*)d_in[3];

    unsigned short* Qb = (unsigned short*)d_ws;                    // [B][N][64] bf16, 2 MB
    unsigned short* Kb = Qb + (size_t)B_ * NSEQ * CKEY;            // [B][N][64] bf16, 2 MB
    unsigned short* Vb = Kb + (size_t)B_ * NSEQ * CKEY;            // [B][128][N] bf16, 4 MB
    float* out = (float*)d_out;

    proj_kernel<<<B_ * (NSEQ / 32), 256, 0, stream>>>(x, Wq, Wk, Wv, Qb, Kb, Vb);
    attn_kernel<<<B_ * (NSEQ / 64), 128, 0, stream>>>(Qb, Kb, Vb, out);
}
